// Classifier_73100343378143
// MI455X (gfx1250) — matrix operand order repaired and checked by reading, not compiled, not run on hardware
//
#include <hip/hip_runtime.h>
#include <hip/hip_bf16.h>

#define N_NODES 50000
#define N_EDGES 800000
#define DIM     128
#define NCLS    10
#define NGRAPH  128
#define NHID    3
#define TILES_PER_BLOCK 5   // 3125 row-tiles of 16 = 625 blocks * 5

typedef __attribute__((ext_vector_type(16))) __bf16 v16bf;
typedef __attribute__((ext_vector_type(8)))  float  v8f;

// ---------------------------------------------------------------- utilities
__global__ void zero_f32(float* __restrict__ p, int n) {
  int i = blockIdx.x * blockDim.x + threadIdx.x;
  if (i < n) p[i] = 0.0f;
}

// degree accumulation via float atomics (counts are exact in f32 up to 2^24)
__global__ void deg_kernel(const int* __restrict__ src, const int* __restrict__ dst,
                           float* __restrict__ outdeg, float* __restrict__ indeg) {
  int e = blockIdx.x * blockDim.x + threadIdx.x;
  if (e < N_EDGES) {
    atomicAdd(&outdeg[src[e]], 1.0f);
    atomicAdd(&indeg[dst[e]], 1.0f);
  }
}

// v <- clip(v,1)^-0.5, applied in place to both degree arrays (contiguous)
__global__ void rsqrt_clip_kernel(float* __restrict__ p, int n) {
  int i = blockIdx.x * blockDim.x + threadIdx.x;
  if (i < n) {
    float v = p[i];
    v = v < 1.0f ? 1.0f : v;
    p[i] = rsqrtf(v);
  }
}

// ------------------------------------------------- sparse aggregation (A*h)
// One wave per edge; lane handles 4 consecutive features (float4 gather,
// 4 global_atomic_add_f32 scatter). c_src scaling fused into the gather.
__global__ void scatter_kernel(const float* __restrict__ h,
                               const float* __restrict__ c_src,
                               const int* __restrict__ src,
                               const int* __restrict__ dst,
                               float* __restrict__ m) {
  long long t = (long long)blockIdx.x * blockDim.x + threadIdx.x;
  int e    = (int)(t >> 5);
  int lane = (int)(t & 31);
  if (e >= N_EDGES) return;
  int s = src[e];
  int d = dst[e];
  float c = c_src[s];
  const float4* hs = (const float4*)(h + (size_t)s * DIM);
  float4 v = hs[lane];
  float* md = m + (size_t)d * DIM + lane * 4;
  atomicAdd(md + 0, v.x * c);
  atomicAdd(md + 1, v.y * c);
  atomicAdd(md + 2, v.z * c);
  atomicAdd(md + 3, v.w * c);
}

// ----------------------------------------- dense transform: relu(cD*m @ W + b)
// Block = 256 threads = 8 waves; wave w owns output column tile n0 = 16*w.
// Each block processes TILES_PER_BLOCK row-tiles of 16 (N = 3125 tiles exact).
// A (f32 agg buffer) converted to bf16 on the fly; W loaded/converted once
// per block into B fragments. c_dst scale + bias + ReLU fused in epilogue.
__global__ __launch_bounds__(256) void gcn_gemm_wmma(
    const float* __restrict__ A,     // [N, 128] aggregated features (f32)
    const float* __restrict__ W,     // [128, 128] row-major
    const float* __restrict__ bias,  // [128]
    const float* __restrict__ c_dst, // [N]
    float* __restrict__ out) {       // [N, 128]
  int wave = threadIdx.x >> 5;
  int lane = threadIdx.x & 31;
  int hi   = (lane >> 4) & 1;
  int col  = wave * 16 + (lane & 15);

  // B fragments: 32x16 bf16 per k-step; lane holds column `col`,
  // K = kb*32 + hi*8 + i (i<8) / kb*32 + 16 + hi*8 + (i-8) (i>=8)
  v16bf bfrag[4];
#pragma unroll
  for (int kb = 0; kb < 4; ++kb) {
    int kbase = kb * 32 + hi * 8;
    v16bf b;
#pragma unroll
    for (int i = 0; i < 16; ++i)
      b[i] = (__bf16)W[(kbase + (i < 8 ? i : i + 8)) * DIM + col];
    bfrag[kb] = b;
  }
  float bv = bias[col];

  int tile0 = blockIdx.x * TILES_PER_BLOCK;
#pragma unroll
  for (int t = 0; t < TILES_PER_BLOCK; ++t) {
    int r0 = (tile0 + t) * 16;
    if (r0 >= N_NODES) return;
    v8f acc = {};
#pragma unroll
    for (int kb = 0; kb < 4; ++kb) {
      // A fragment 16x32 bf16: lane row = r0 + lane%16,
      // K = klo + i (i<8) / klo + 8 + i (i>=8), klo = kb*32 + hi*8
      int klo = kb * 32 + hi * 8;
      const float* row = A + (size_t)(r0 + (lane & 15)) * DIM + klo;
      v16bf a;
#pragma unroll
      for (int i = 0; i < 8; ++i) a[i] = (__bf16)row[i];
#pragma unroll
      for (int i = 0; i < 8; ++i) a[8 + i] = (__bf16)row[16 + i];
      acc = __builtin_amdgcn_wmma_f32_16x16x32_bf16(
          /*neg_a=*/false, a, /*neg_b=*/false, bfrag[kb],
          /*c_mod=*/(short)0, acc, /*reuse_a=*/false, /*reuse_b=*/false);
    }
    // Epilogue: D VGPR j holds row (r0 + j + hi*8), column `col`
#pragma unroll
    for (int j = 0; j < 8; ++j) {
      int row = r0 + j + hi * 8;
      float v = acc[j] * c_dst[row] + bv;
      out[(size_t)row * DIM + col] = v > 0.0f ? v : 0.0f;
    }
  }
}

// ------------------------------------------------------------- readout
__global__ void readout_kernel(const float* __restrict__ h,
                               const int* __restrict__ gid,
                               float* __restrict__ hg,   // [G,128] sums
                               float* __restrict__ cnt) {// [G]
  long long t = (long long)blockIdx.x * blockDim.x + threadIdx.x;
  int n    = (int)(t >> 5);
  int lane = (int)(t & 31);
  if (n >= N_NODES) return;
  int g = gid[n];
  const float4* hn = (const float4*)(h + (size_t)n * DIM);
  float4 v = hn[lane];
  float* p = hg + (size_t)g * DIM + lane * 4;
  atomicAdd(p + 0, v.x);
  atomicAdd(p + 1, v.y);
  atomicAdd(p + 2, v.z);
  atomicAdd(p + 3, v.w);
  if (lane == 0) atomicAdd(&cnt[g], 1.0f);
}

__global__ void final_kernel(const float* __restrict__ hg,
                             const float* __restrict__ cnt,
                             const float* __restrict__ Wc,  // [128,10]
                             const float* __restrict__ bc,  // [10]
                             float* __restrict__ out) {     // [G,10]
  int t = blockIdx.x * blockDim.x + threadIdx.x;
  if (t >= NGRAPH * NCLS) return;
  int g = t / NCLS;
  int c = t % NCLS;
  float inv = 1.0f / fmaxf(cnt[g], 1.0f);
  float s = 0.0f;
#pragma unroll 8
  for (int k = 0; k < DIM; ++k)
    s += hg[(size_t)g * DIM + k] * Wc[k * NCLS + c];
  out[t] = s * inv + bc[c];
}

// ---------------------------------------------------------------- driver
extern "C" void kernel_launch(void* const* d_in, const int* in_sizes, int n_in,
                              void* d_out, int out_size, void* d_ws, size_t ws_size,
                              hipStream_t stream) {
  const float* x    = (const float*)d_in[0];  // [N,128]
  const float* W0   = (const float*)d_in[1];  // [128,128]
  const float* b0   = (const float*)d_in[2];  // [128]
  const float* Ws   = (const float*)d_in[3];  // [3,128,128]
  const float* bs   = (const float*)d_in[4];  // [3,128]
  const float* Wc   = (const float*)d_in[5];  // [128,10]
  const float* bc   = (const float*)d_in[6];  // [10]
  const int*   eidx = (const int*)d_in[7];    // [2,E] row-major
  const int*   gid  = (const int*)d_in[8];    // [N]
  const int*   src  = eidx;                   // row 0
  const int*   dst  = eidx + N_EDGES;         // row 1
  float* out = (float*)d_out;

  char* ws = (char*)d_ws;
  float* c_src = (float*)ws;  ws += (size_t)N_NODES * 4;
  float* c_dst = (float*)ws;  ws += (size_t)N_NODES * 4;
  float* hA    = (float*)ws;  ws += (size_t)N_NODES * DIM * 4;
  float* hB    = (float*)ws;  ws += (size_t)N_NODES * DIM * 4;
  float* mbuf  = (float*)ws;  ws += (size_t)N_NODES * DIM * 4;
  float* hg    = (float*)ws;  ws += (size_t)NGRAPH * DIM * 4;
  float* cnt   = (float*)ws;  ws += (size_t)NGRAPH * 4;

  const int T = 256;
  // --- degree normalization coefficients ---
  zero_f32<<<(2 * N_NODES + T - 1) / T, T, 0, stream>>>(c_src, 2 * N_NODES);
  deg_kernel<<<(N_EDGES + T - 1) / T, T, 0, stream>>>(src, dst, c_src, c_dst);
  rsqrt_clip_kernel<<<(2 * N_NODES + T - 1) / T, T, 0, stream>>>(c_src, 2 * N_NODES);

  const int feat_elems = N_NODES * DIM;
  const int scat_blocks = (int)(((long long)N_EDGES * 32 + T - 1) / T);
  const int gemm_blocks = (N_NODES / 16 + TILES_PER_BLOCK - 1) / TILES_PER_BLOCK;

  // --- 4 GraphConv layers ---
  const float* hin = x;
  float* hout = hA;
  for (int layer = 0; layer < 1 + NHID; ++layer) {
    const float* W = (layer == 0) ? W0 : (Ws + (size_t)(layer - 1) * DIM * DIM);
    const float* b = (layer == 0) ? b0 : (bs + (size_t)(layer - 1) * DIM);
    zero_f32<<<(feat_elems + T - 1) / T, T, 0, stream>>>(mbuf, feat_elems);
    scatter_kernel<<<scat_blocks, T, 0, stream>>>(hin, c_src, src, dst, mbuf);
    gcn_gemm_wmma<<<gemm_blocks, T, 0, stream>>>(mbuf, W, b, c_dst, hout);
    hin = hout;
    hout = (hout == hA) ? hB : hA;
  }

  // --- mean readout + classifier ---
  zero_f32<<<(NGRAPH * DIM + NGRAPH + T - 1) / T, T, 0, stream>>>(hg, NGRAPH * DIM + NGRAPH);
  readout_kernel<<<(int)(((long long)N_NODES * 32 + T - 1) / T), T, 0, stream>>>(hin, gid, hg, cnt);
  final_kernel<<<(NGRAPH * NCLS + T - 1) / T, T, 0, stream>>>(hg, cnt, Wc, bc, out);
}
